// SampleContrastiveLoss_77653008711742
// MI455X (gfx1250) — matrix operand order repaired and checked by reading, not compiled, not run on hardware
//
#include <hip/hip_runtime.h>
#include <hip/hip_bf16.h>
#include <math.h>

// ---------------------------------------------------------------------------
// SampleContrastiveLoss on MI455X (gfx1250, wave32, WMMA)
//
// N = 16384 rows, D = 128.  sim = n0 @ n1^T / T  (68.7 GFLOP) -> compute
// bound => v_wmma_f32_16x16x32_f16, streamed (never materialize NxN):
//   denom_i = sum_j mask_j * exp(sim_ij),  num_i = exp(sim_ii)
//   loss = sum_i mask_i * -log((num_i+1e-8)/(denom_i+1e-8)) / sum(mask)
//
// Round 5 changes:
//  * cap the j-tile loop at unroll 2: round 4's full unroll software-
//    pipelined ~24 ds_load_b128 in flight and spilled past 256 VGPRs
//    (s_set_vgpr_msb toggles + v_dual_mov shuffling in the hot loop,
//    ~3 waves/SIMD).  16 WMMAs/iter across 4 independent chains is still
//    ample ILP; live set halves and allocation returns under 256 VGPRs.
// ---------------------------------------------------------------------------

typedef __attribute__((ext_vector_type(16))) _Float16 v16h;
typedef __attribute__((ext_vector_type(8)))  float    v8f;
typedef __attribute__((ext_vector_type(4)))  _Float16 v4h;

#define NROWS   16384
#define DDIM    128
#define INV_T   (1.0f / 0.07f)
#define LOG2E   1.4426950408889634f
#define C2SCALE (INV_T * LOG2E)       // exp(x/T) == exp2(x * C2SCALE)
#define EPS_N   1e-12f
#define EPS_L   1e-8f
#define JCHUNK  64                    // j-rows per LDS buffer (16KB)
#define JHALF   (NROWS / 2)

union V16U { v16h v; uint4 q[2]; };

// ---------------------------------------------------------------------------
// Kernel 1: per row r — L2-normalize fea0[r], fea1[r]; store f16 rows
// (n0h pre-scaled by C2SCALE); numerator num[r] = exp2(C2 * <n0,n1>) in f32.
// One wave per row.
// ---------------------------------------------------------------------------
__global__ void k_norm_num(const float* __restrict__ f0,
                           const float* __restrict__ f1,
                           _Float16* __restrict__ n0h,
                           _Float16* __restrict__ n1h,
                           float* __restrict__ num) {
    const int lane   = threadIdx.x & 31;
    const int gwave  = (blockIdx.x * blockDim.x + threadIdx.x) >> 5;
    const int nwaves = (gridDim.x * blockDim.x) >> 5;

    for (int r = gwave; r < NROWS; r += nwaves) {
        float4 x = *(const float4*)(f0 + (size_t)r * DDIM + lane * 4);
        float4 y = *(const float4*)(f1 + (size_t)r * DDIM + lane * 4);
        float ss0 = x.x * x.x + x.y * x.y + x.z * x.z + x.w * x.w;
        float ss1 = y.x * y.x + y.y * y.y + y.z * y.z + y.w * y.w;
        #pragma unroll
        for (int off = 16; off; off >>= 1) {
            ss0 += __shfl_xor(ss0, off, 32);
            ss1 += __shfl_xor(ss1, off, 32);
        }
        float inv0 = 1.0f / fmaxf(sqrtf(ss0), EPS_N);
        float inv1 = 1.0f / fmaxf(sqrtf(ss1), EPS_N);

        // A operand scaled by C2SCALE (folds the exp2 argument scale).
        v4h h0, h1;
        h0[0] = (_Float16)(x.x * inv0 * C2SCALE);
        h0[1] = (_Float16)(x.y * inv0 * C2SCALE);
        h0[2] = (_Float16)(x.z * inv0 * C2SCALE);
        h0[3] = (_Float16)(x.w * inv0 * C2SCALE);
        h1[0] = (_Float16)(y.x * inv1);
        h1[1] = (_Float16)(y.y * inv1);
        h1[2] = (_Float16)(y.z * inv1);
        h1[3] = (_Float16)(y.w * inv1);
        *(v4h*)(n0h + (size_t)r * DDIM + lane * 4) = h0;
        *(v4h*)(n1h + (size_t)r * DDIM + lane * 4) = h1;

        // Numerator: exp(<n0,n1>/T), f32 path (diagonal of sim).
        float d = (x.x * inv0) * (y.x * inv1) + (x.y * inv0) * (y.y * inv1)
                + (x.z * inv0) * (y.z * inv1) + (x.w * inv0) * (y.w * inv1);
        #pragma unroll
        for (int off = 16; off; off >>= 1) d += __shfl_xor(d, off, 32);
        if (lane == 0) num[r] = __builtin_amdgcn_exp2f(d * C2SCALE);
    }
}

// ---------------------------------------------------------------------------
// Kernel 2: fused WMMA GEMM + exp + masked row-sum (denominator only).
// Grid: (i-block 0..127) x (j-half 0..1).  Block = 128 threads = 4 waves;
// wave w owns i-rows [ib*128 + 32w, +32) as two 16-row A tiles.
// j streamed in 64-row chunks, double-buffered in LDS via async loads.
// ---------------------------------------------------------------------------
__global__ void __launch_bounds__(128)
k_sim_reduce(const _Float16* __restrict__ n0h,
             const _Float16* __restrict__ n1h,
             const int*      __restrict__ mask,
             float* __restrict__ denOut) {   // [2][NROWS] partials
    __shared__ _Float16 lds_b[2][JCHUNK * DDIM];   // 2 x 16 KB
    __shared__ float    lds_m[2][JCHUNK];

    const int tid  = threadIdx.x;
    const int wave = tid >> 5;
    const int lane = tid & 31;
    const int h    = lane >> 4;        // lane half (ISA A/B layout selector)
    const int n16  = lane & 15;        // row (A) / col (B,C,D) index in tile
    const int jh   = blockIdx.x & 1;   // j-half
    const int ib   = blockIdx.x >> 1;  // i-block
    const int i0   = ib * 128 + wave * 32;   // first of this wave's 32 rows
    const int jbeg = jh * JHALF;

    // Issue async global->LDS copy of one 64-row chunk (ASYNCcnt-tracked).
    auto stage_async = [&](int jc, int buf) {
        const unsigned long long gbase =
            (unsigned long long)(const void*)(n1h + (size_t)jc * DDIM);
        const unsigned lbase = (unsigned)(size_t)(void*)&lds_b[buf][0];
        #pragma unroll
        for (int q = 0; q < 8; ++q) {
            unsigned off = (unsigned)(q * 128 + tid) * 16u;
            unsigned ldst = lbase + off;
            unsigned long long ga = gbase + off;
            asm volatile("global_load_async_to_lds_b128 %0, %1, off"
                         :: "v"(ldst), "v"(ga) : "memory");
        }
        if (tid < JCHUNK) lds_m[buf][tid] = mask[jc + tid] ? 1.0f : 0.0f;
    };

    // A fragments (loop invariant), two 16-row tiles; per ISA 16-bit A layout:
    // halves[0..7] = K = k0 + 8h + j,  halves[8..15] = K = k0 + 16 + 8h + j.
    v16h afrA[4], afrB[4];
    {
        const _Float16* arowA = n0h + (size_t)(i0 + n16) * DDIM;
        const _Float16* arowB = n0h + (size_t)(i0 + 16 + n16) * DDIM;
        #pragma unroll
        for (int k = 0; k < 4; ++k) {
            V16U ua, ub;
            ua.q[0] = *(const uint4*)(arowA + k * 32 + h * 8);
            ua.q[1] = *(const uint4*)(arowA + k * 32 + 16 + h * 8);
            ub.q[0] = *(const uint4*)(arowB + k * 32 + h * 8);
            ub.q[1] = *(const uint4*)(arowB + k * 32 + 16 + h * 8);
            afrA[k] = ua.v;
            afrB[k] = ub.v;
        }
    }

    float dsumA[8], dsumB[8];
    #pragma unroll
    for (int v = 0; v < 8; ++v) { dsumA[v] = 0.0f; dsumB[v] = 0.0f; }

    const int nch = JHALF / JCHUNK;

    stage_async(jbeg, 0);
    asm volatile("s_wait_asynccnt 0x0" ::: "memory");
    __syncthreads();

    for (int c = 0; c < nch; ++c) {
        const int cur = c & 1;
        if (c + 1 < nch) stage_async(jbeg + (c + 1) * JCHUNK, cur ^ 1);

        #pragma unroll 2
        for (int jt = 0; jt < JCHUNK / 16; ++jt) {
            // B fragment: col = n16; halves[0..7] = K = k0+8h+j, halves[8..15] = K = k0+16+8h+j.
            // One B fragment feeds both A tiles (two independent WMMA chains).
            v8f accA = {}, accB = {};
            #pragma unroll
            for (int k = 0; k < 4; ++k) {
                V16U u;
                const _Float16* bsrc =
                    &lds_b[cur][(jt * 16 + n16) * DDIM + k * 32 + h * 8];
                u.q[0] = *(const uint4*)(bsrc);
                u.q[1] = *(const uint4*)(bsrc + 16);
                accA = __builtin_amdgcn_wmma_f32_16x16x32_f16(
                        false, afrA[k], false, u.v, (short)0, accA, false, false);
                accB = __builtin_amdgcn_wmma_f32_16x16x32_f16(
                        false, afrB[k], false, u.v, (short)0, accB, false, false);
            }
            // acc[v] = C2SCALE * sim for element (M = v + 8h, N = n16).
            const float msk = lds_m[cur][jt * 16 + n16];
            #pragma unroll
            for (int v = 0; v < 8; ++v) {
                dsumA[v] += __builtin_amdgcn_exp2f(accA[v]) * msk;
                dsumB[v] += __builtin_amdgcn_exp2f(accB[v]) * msk;
            }
        }

        asm volatile("s_wait_asynccnt 0x0" ::: "memory");
        __syncthreads();
    }

    // Row sums: reduce across the 16 lanes of each half (same M, all N).
    float* dpart = denOut + (size_t)jh * NROWS;
    #pragma unroll
    for (int v = 0; v < 8; ++v) {
        float sA = dsumA[v], sB = dsumB[v];
        #pragma unroll
        for (int off = 8; off; off >>= 1) {
            sA += __shfl_xor(sA, off, 16);
            sB += __shfl_xor(sB, off, 16);
        }
        if (n16 == 0) {
            dpart[i0 + 8 * h + v]      = sA;
            dpart[i0 + 16 + 8 * h + v] = sB;
        }
    }
}

// ---------------------------------------------------------------------------
// Kernel 3: final masked loss reduction (single block, deterministic).
// ---------------------------------------------------------------------------
__global__ void k_finalize(const float* __restrict__ num,
                           const float* __restrict__ den,   // [2][NROWS]
                           const int*   __restrict__ mask,
                           float* __restrict__ out) {
    __shared__ float s_loss[256];
    __shared__ float s_cnt[256];
    float l = 0.0f, c = 0.0f;
    for (int i = threadIdx.x; i < NROWS; i += 256) {
        if (mask[i]) {
            float d = den[i] + den[NROWS + i];
            l += -logf((num[i] + EPS_L) / (d + EPS_L));
            c += 1.0f;
        }
    }
    s_loss[threadIdx.x] = l;
    s_cnt[threadIdx.x]  = c;
    __syncthreads();
    for (int off = 128; off; off >>= 1) {
        if (threadIdx.x < off) {
            s_loss[threadIdx.x] += s_loss[threadIdx.x + off];
            s_cnt[threadIdx.x]  += s_cnt[threadIdx.x + off];
        }
        __syncthreads();
    }
    if (threadIdx.x == 0) out[0] = s_loss[0] / s_cnt[0];
}

// ---------------------------------------------------------------------------
extern "C" void kernel_launch(void* const* d_in, const int* in_sizes, int n_in,
                              void* d_out, int out_size, void* d_ws, size_t ws_size,
                              hipStream_t stream) {
    const float* fea0 = (const float*)d_in[0];
    const float* fea1 = (const float*)d_in[1];
    const int*   mask = (const int*)d_in[2];
    float* out = (float*)d_out;

    // Workspace layout: n0h[4MB] | n1h[4MB] | num[64KB] | den[2][64KB]
    char* ws = (char*)d_ws;
    _Float16* n0h = (_Float16*)(ws);
    _Float16* n1h = (_Float16*)(ws + (size_t)NROWS * DDIM * 2);
    float*    num = (float*)(ws + (size_t)NROWS * DDIM * 4);
    float*    den = (float*)(ws + (size_t)NROWS * DDIM * 4 + NROWS * 4);

    k_norm_num<<<512, 256, 0, stream>>>(fea0, fea1, n0h, n1h, num);
    k_sim_reduce<<<256, 128, 0, stream>>>(n0h, n1h, mask, den);
    k_finalize<<<1, 256, 0, stream>>>(num, den, mask, out);
}
